// TritonFusedRCMAttention_15126874816819
// MI455X (gfx1250) — matrix operand order repaired and checked by reading, not compiled, not run on hardware
//
#include <hip/hip_runtime.h>

// ---------------------------------------------------------------------------
// Types / WMMA helpers (gfx1250, wave32)
// ---------------------------------------------------------------------------
typedef __bf16 bf16_t;
typedef __attribute__((ext_vector_type(16))) __bf16 v16bf;
typedef __attribute__((ext_vector_type(8)))  __bf16 v8bf;
typedef __attribute__((ext_vector_type(4)))  __bf16 v4bf;
typedef __attribute__((ext_vector_type(8)))  float  v8f;

// Pointer types for the async global->LDS copy builtin (per hipcc diagnostic:
// params are int4-vector pointers, global (AS1) source, LDS (AS3) dest).
typedef int v4i __attribute__((__vector_size__(4 * sizeof(int))));
typedef __attribute__((address_space(1))) v4i as1_v4i;
typedef __attribute__((address_space(3))) v4i as3_v4i;

#if defined(__has_builtin)
#if __has_builtin(__builtin_amdgcn_global_load_async_to_lds_b128)
#define HAVE_ASYNC_COPY 1
#endif
#endif
#ifndef HAVE_ASYNC_COPY
#define HAVE_ASYNC_COPY 0
#endif

#define WMMA_BF16(a, b, c) \
  __builtin_amdgcn_wmma_f32_16x16x32_bf16(false, (a), false, (b), (short)0, (c), false, false)

static __device__ __forceinline__ v8f zero8() {
  v8f z = {0.f, 0.f, 0.f, 0.f, 0.f, 0.f, 0.f, 0.f};
  return z;
}

static __device__ __forceinline__ v16bf cat8(v8bf lo, v8bf hi) {
  v16bf r;
#pragma unroll
  for (int i = 0; i < 8; ++i) { r[i] = lo[i]; r[i + 8] = hi[i]; }
  return r;
}

// A fragment, 16x32 bf16 row-major source. Lane: m = lane&15, half = lane>>4.
// elems 0..7  = A[m][k0 + half*8 + e] ; elems 8..15 = A[m][k0 + 16 + half*8 + e]
static __device__ __forceinline__ v16bf load_a_frag(const bf16_t* row_k0, int half) {
  v8bf lo = *(const v8bf*)(row_k0 + half * 8);
  v8bf hi = *(const v8bf*)(row_k0 + 16 + half * 8);
  return cat8(lo, hi);
}

// B fragment, 32x16 bf16 row-major source. Lane: n = lane&15, half = lane>>4.
// elems 0..7 = B[k0 + half*8 + e][n] ; elems 8..15 = B[k0 + 16 + half*8 + e][n]
static __device__ __forceinline__ v16bf load_b_frag(const bf16_t* col_k0, int ld, int half) {
  v16bf r;
#pragma unroll
  for (int e = 0; e < 8; ++e) {
    r[e] = col_k0[(size_t)(half * 8 + e) * ld];
    r[e + 8] = col_k0[(size_t)(16 + half * 8 + e) * ld];
  }
  return r;
}

// 32-byte global -> LDS stage: async DMA (ASYNCcnt) when available.
static __device__ __forceinline__ void stage32B(const bf16_t* __restrict__ g,
                                                bf16_t* l) {
#if HAVE_ASYNC_COPY
  __builtin_amdgcn_global_load_async_to_lds_b128((as1_v4i*)(g), (as3_v4i*)(l), 0, 0);
  __builtin_amdgcn_global_load_async_to_lds_b128((as1_v4i*)(g), (as3_v4i*)(l), 16, 0);
#else
  const uint4* s = (const uint4*)g;
  ((uint4*)l)[0] = s[0];
  ((uint4*)l)[1] = s[1];
#endif
}

static __device__ __forceinline__ void wait_async() {
#if HAVE_ASYNC_COPY
#if __has_builtin(__builtin_amdgcn_s_wait_asynccnt)
  __builtin_amdgcn_s_wait_asynccnt(0);
#else
  asm volatile("s_wait_asynccnt 0" ::: "memory");
#endif
#endif
}

// ---------------------------------------------------------------------------
// Problem constants
// ---------------------------------------------------------------------------
#define BATCH 4
#define NSEQ  2048
#define C_DIM 512
#define HEADS 8
#define HD    64

// ---------------------------------------------------------------------------
// fp32 -> bf16 conversion (4 elements / thread)
// ---------------------------------------------------------------------------
__global__ void cvt_f32_to_bf16_kernel(const float* __restrict__ in,
                                       bf16_t* __restrict__ out, int n4) {
  const int i = blockIdx.x * blockDim.x + threadIdx.x;
  if (i >= n4) return;
  const float4 v = ((const float4*)in)[i];
  v4bf o = {(bf16_t)v.x, (bf16_t)v.y, (bf16_t)v.z, (bf16_t)v.w};
  *(v4bf*)(out + 4 * (size_t)i) = o;
}

// ---------------------------------------------------------------------------
// WMMA main loop: one wave computes a 16x64 tile, K-chunks of 32.
// Software-pipelined: operands for chunk k+1 are loaded (register double
// buffer) before the WMMAs of chunk k issue, so loads overlap the matrix pipe.
// ---------------------------------------------------------------------------
static __device__ __forceinline__ void gemm_tile_16x64(
    const bf16_t* __restrict__ A, int lda, const bf16_t* __restrict__ B,
    int ldb, int K, int m_base, int n_base, int lane, v8f acc[4]) {
  const int lm = lane & 15, half = lane >> 4;
  const bf16_t* arow = A + (size_t)(m_base + lm) * lda;
  const bf16_t* brow = B + n_base + lm;

  v16bf af  = load_a_frag(arow, half);
  v16bf bf0 = load_b_frag(brow, ldb, half);
  v16bf bf1 = load_b_frag(brow + 16, ldb, half);
  v16bf bf2 = load_b_frag(brow + 32, ldb, half);
  v16bf bf3 = load_b_frag(brow + 48, ldb, half);

  for (int k0 = 32; k0 < K; k0 += 32) {
    const bf16_t* bn = brow + (size_t)k0 * ldb;
    v16bf afn = load_a_frag(arow + k0, half);
    v16bf bn0 = load_b_frag(bn, ldb, half);
    v16bf bn1 = load_b_frag(bn + 16, ldb, half);
    v16bf bn2 = load_b_frag(bn + 32, ldb, half);
    v16bf bn3 = load_b_frag(bn + 48, ldb, half);
    acc[0] = WMMA_BF16(af, bf0, acc[0]);
    acc[1] = WMMA_BF16(af, bf1, acc[1]);
    acc[2] = WMMA_BF16(af, bf2, acc[2]);
    acc[3] = WMMA_BF16(af, bf3, acc[3]);
    af = afn; bf0 = bn0; bf1 = bn1; bf2 = bn2; bf3 = bn3;
  }
  acc[0] = WMMA_BF16(af, bf0, acc[0]);
  acc[1] = WMMA_BF16(af, bf1, acc[1]);
  acc[2] = WMMA_BF16(af, bf2, acc[2]);
  acc[3] = WMMA_BF16(af, bf3, acc[3]);
}

// ---------------------------------------------------------------------------
// QKV projection: [8192,512]x[512,1536]+bias, scatter to Q / K^T / V (bf16)
// grid (6, 256), block 256 (8 waves; wave -> 16x64 tile; block -> 32x256)
// ---------------------------------------------------------------------------
__global__ __launch_bounds__(256) void qkv_gemm_kernel(
    const bf16_t* __restrict__ xb, const bf16_t* __restrict__ wb,
    const float* __restrict__ bias, bf16_t* __restrict__ qg,
    bf16_t* __restrict__ ktg, bf16_t* __restrict__ vg) {
  const int lane = threadIdx.x & 31, wave = threadIdx.x >> 5;
  const int m_base = blockIdx.y * 32 + (wave & 1) * 16;
  const int n_base = blockIdx.x * 256 + (wave >> 1) * 64;
  v8f acc[4] = {zero8(), zero8(), zero8(), zero8()};
  gemm_tile_16x64(xb, C_DIM, wb, 3 * C_DIM, C_DIM, m_base, n_base, lane, acc);

  const int lm = lane & 15, half = lane >> 4;
#pragma unroll
  for (int t = 0; t < 4; ++t) {
    const int j = n_base + t * 16 + lm;            // 0..1535
    const float bj = bias[j];
    const int three = j >> 9, h = (j >> 6) & 7, d = j & 63;
#pragma unroll
    for (int r = 0; r < 8; ++r) {
      const int mrow = m_base + r + 8 * half;      // 0..8191
      const int b = mrow >> 11, n = mrow & (NSEQ - 1);
      const bf16_t val = (bf16_t)(acc[t][r] + bj);
      const size_t bh = (size_t)(b * HEADS + h);
      if (three == 0)      qg [(bh * NSEQ + n) * HD + d] = val;
      else if (three == 1) ktg[(bh * HD + d) * NSEQ + n] = val;  // transposed
      else                 vg [(bh * NSEQ + n) * HD + d] = val;
    }
  }
}

// ---------------------------------------------------------------------------
// Flash attention + rCM epilogue:  y = c_skip*x + c_out*softmax(QK^T/8)V
// grid 512 (= B*H*16), block 256: 8 waves x 16 query rows, key chunks of 64.
// K^T/V chunks double-buffered in LDS, filled by async global->LDS copies.
// ---------------------------------------------------------------------------
__global__ __launch_bounds__(256) void attn_kernel(
    const bf16_t* __restrict__ qg, const bf16_t* __restrict__ ktg,
    const bf16_t* __restrict__ vg, const float* __restrict__ x,
    const int* __restrict__ tptr, bf16_t* __restrict__ yb) {
  __shared__ __align__(16) bf16_t ktile[2][64 * 64];   // [d][n]  16 KB
  __shared__ __align__(16) bf16_t vtile[2][64 * 64];   // [n][d]  16 KB
  __shared__ __align__(16) bf16_t ptile[8][16 * 64];   // per-wave 16 KB

  const int lane = threadIdx.x & 31, wave = threadIdx.x >> 5;
  const int lm = lane & 15, half = lane >> 4;
  const int bh = blockIdx.x >> 4;                      // b*H + h
  const int q0 = (blockIdx.x & 15) * 128 + wave * 16;  // first query row

  // Q fragments kept in registers for the whole kernel
  const bf16_t* qrow = qg + ((size_t)bh * NSEQ + q0 + lm) * HD;
  const v16bf qf0 = load_a_frag(qrow + 0, half);
  const v16bf qf1 = load_a_frag(qrow + 32, half);

  v8f o[4] = {zero8(), zero8(), zero8(), zero8()};
  float mrun[8], lrun[8];
#pragma unroll
  for (int r = 0; r < 8; ++r) { mrun[r] = -3.0e38f; lrun[r] = 0.f; }

  const float sl = 0.125f * 1.44269504088896340736f;   // D^-0.5 * log2(e)
  const bf16_t* kbase = ktg + (size_t)bh * HD * NSEQ;  // [d][n]
  const bf16_t* vbase = vg + (size_t)bh * NSEQ * HD;   // [n][d]
  bf16_t* prow = &ptile[wave][0];

  // staging thread mapping: 32 contiguous bytes per thread per tile
  const int sd = threadIdx.x >> 2;                     // 0..63
  const int sn = (threadIdx.x & 3) * 16;               // 0,16,32,48

  // prologue: stage chunk 0 into buffer 0
  stage32B(kbase + (size_t)sd * NSEQ + sn, &ktile[0][sd * 64 + sn]);
  stage32B(vbase + (size_t)sd * HD + sn, &vtile[0][sd * 64 + sn]);

  for (int ic = 0; ic < NSEQ / 64; ++ic) {
    const int kc = ic * 64;
    const int cur = ic & 1;
    wait_async();        // own async copies (current buffer) done
    __syncthreads();     // everyone's copies visible; prev buffer free

    if (ic + 1 < NSEQ / 64) {  // stage next chunk into the other buffer
      const int nkc = kc + 64, nb = cur ^ 1;
      stage32B(kbase + (size_t)sd * NSEQ + nkc + sn, &ktile[nb][sd * 64 + sn]);
      stage32B(vbase + (size_t)(nkc + sd) * HD + sn, &vtile[nb][sd * 64 + sn]);
#if !HAVE_ASYNC_COPY
      if (nkc + 64 < NSEQ) {  // global_prefetch_b8 of the chunk after that
        __builtin_prefetch(kbase + (size_t)sd * NSEQ + nkc + 64 + sn, 0, 1);
        __builtin_prefetch(vbase + (size_t)(nkc + 64 + sd) * HD + sn, 0, 1);
      }
#endif
    }
    const bf16_t* kt = &ktile[cur][0];
    const bf16_t* vt = &vtile[cur][0];

    // ---- scores S[16x64] = Q * K^T (4 tiles, K-dim = 64 in two chunks) ----
    v8f s[4];
#pragma unroll
    for (int t = 0; t < 4; ++t) {
      v8f a_ = zero8();
      a_ = WMMA_BF16(qf0, load_b_frag(&kt[0 * 64 + t * 16 + lm], 64, half), a_);
      a_ = WMMA_BF16(qf1, load_b_frag(&kt[32 * 64 + t * 16 + lm], 64, half), a_);
      s[t] = a_;
    }

    // ---- online softmax (row reductions across 16-lane groups) ----
    float mx[8];
#pragma unroll
    for (int r = 0; r < 8; ++r)
      mx[r] = fmaxf(fmaxf(s[0][r], s[1][r]), fmaxf(s[2][r], s[3][r]));
#pragma unroll
    for (int off = 1; off < 16; off <<= 1)
#pragma unroll
      for (int r = 0; r < 8; ++r)
        mx[r] = fmaxf(mx[r], __shfl_xor(mx[r], off, 32));

    float corr[8], rs[8];
#pragma unroll
    for (int r = 0; r < 8; ++r) {
      const float mn = fmaxf(mrun[r], mx[r] * sl);
      corr[r] = __builtin_amdgcn_exp2f(mrun[r] - mn);
      mrun[r] = mn;
      rs[r] = 0.f;
    }
#pragma unroll
    for (int t = 0; t < 4; ++t)
#pragma unroll
      for (int r = 0; r < 8; ++r) {
        const float p = __builtin_amdgcn_exp2f(s[t][r] * sl - mrun[r]);
        rs[r] += p;
        // C/D layout -> row-major P in wave-private LDS (A-layout source)
        prow[(r + 8 * half) * 64 + t * 16 + lm] = (bf16_t)p;
      }
#pragma unroll
    for (int off = 1; off < 16; off <<= 1)
#pragma unroll
      for (int r = 0; r < 8; ++r)
        rs[r] += __shfl_xor(rs[r], off, 32);
#pragma unroll
    for (int r = 0; r < 8; ++r) {
      lrun[r] = lrun[r] * corr[r] + rs[r];
#pragma unroll
      for (int t = 0; t < 4; ++t) o[t][r] *= corr[r];
    }

    // ---- O += P * V (kn chunks of 32) ----
#pragma unroll
    for (int nc = 0; nc < 64; nc += 32) {
      const v16bf pf = load_a_frag(prow + lm * 64 + nc, half);
#pragma unroll
      for (int t = 0; t < 4; ++t)
        o[t] = WMMA_BF16(pf, load_b_frag(&vt[nc * 64 + t * 16 + lm], 64, half), o[t]);
    }
  }

  // ---- normalize + rCM residual epilogue ----
  const float tf = (float)tptr[0];
  const float cskip = 1.0f / (tf * tf + 1.0f);               // sigma_data = 1
  const float cout = tf * rsqrtf(tf * tf + 1.0f);
  const int b = bh >> 3, h = bh & 7;
  float inv[8];
#pragma unroll
  for (int r = 0; r < 8; ++r) inv[r] = 1.0f / lrun[r];
#pragma unroll
  for (int t = 0; t < 4; ++t)
#pragma unroll
    for (int r = 0; r < 8; ++r) {
      const int nrow = q0 + r + 8 * half;
      const int c = h * HD + t * 16 + lm;
      const size_t idx = ((size_t)(b * NSEQ + nrow)) * C_DIM + c;
      yb[idx] = (bf16_t)(cskip * x[idx] + cout * (o[t][r] * inv[r]));
    }
}

// ---------------------------------------------------------------------------
// Output projection: out = y @ w_proj + b_proj  (f32 out)
// grid (2, 256), block 256
// ---------------------------------------------------------------------------
__global__ __launch_bounds__(256) void proj_gemm_kernel(
    const bf16_t* __restrict__ yb, const bf16_t* __restrict__ wb,
    const float* __restrict__ bias, float* __restrict__ out) {
  const int lane = threadIdx.x & 31, wave = threadIdx.x >> 5;
  const int m_base = blockIdx.y * 32 + (wave & 1) * 16;
  const int n_base = blockIdx.x * 256 + (wave >> 1) * 64;
  v8f acc[4] = {zero8(), zero8(), zero8(), zero8()};
  gemm_tile_16x64(yb, C_DIM, wb, C_DIM, C_DIM, m_base, n_base, lane, acc);

  const int lm = lane & 15, half = lane >> 4;
#pragma unroll
  for (int t = 0; t < 4; ++t) {
    const int j = n_base + t * 16 + lm;
    const float bj = bias[j];
#pragma unroll
    for (int r = 0; r < 8; ++r) {
      const int mrow = m_base + r + 8 * half;
      out[(size_t)mrow * C_DIM + j] = acc[t][r] + bj;
    }
  }
}

// ---------------------------------------------------------------------------
// Launch
// ---------------------------------------------------------------------------
extern "C" void kernel_launch(void* const* d_in, const int* in_sizes, int n_in,
                              void* d_out, int out_size, void* d_ws,
                              size_t ws_size, hipStream_t stream) {
  (void)in_sizes; (void)n_in; (void)out_size; (void)ws_size;
  const float* x      = (const float*)d_in[0];
  const float* w_qkv  = (const float*)d_in[1];
  const float* b_qkv  = (const float*)d_in[2];
  const float* w_proj = (const float*)d_in[3];
  const float* b_proj = (const float*)d_in[4];
  const int*   tptr   = (const int*)d_in[5];

  const size_t XN = (size_t)BATCH * NSEQ * C_DIM;       // 4,194,304
  const size_t WQ = (size_t)C_DIM * 3 * C_DIM;          //   786,432
  const size_t WP = (size_t)C_DIM * C_DIM;              //   262,144
  const size_t QK = (size_t)BATCH * HEADS * NSEQ * HD;  // 4,194,304

  bf16_t* p   = (bf16_t*)d_ws;
  bf16_t* xb  = p; p += XN;
  bf16_t* wqb = p; p += WQ;
  bf16_t* wpb = p; p += WP;
  bf16_t* qg  = p; p += QK;
  bf16_t* ktg = p; p += QK;
  bf16_t* vg  = p; p += QK;
  bf16_t* yb  = p; p += XN;   // total ~42 MB of workspace

  cvt_f32_to_bf16_kernel<<<(int)(XN / 1024), 256, 0, stream>>>(x, xb, (int)(XN / 4));
  cvt_f32_to_bf16_kernel<<<(int)(WQ / 1024), 256, 0, stream>>>(w_qkv, wqb, (int)(WQ / 4));
  cvt_f32_to_bf16_kernel<<<(int)(WP / 1024), 256, 0, stream>>>(w_proj, wpb, (int)(WP / 4));
  qkv_gemm_kernel<<<dim3(6, 256), 256, 0, stream>>>(xb, wqb, b_qkv, qg, ktg, vg);
  attn_kernel<<<512, 256, 0, stream>>>(qg, ktg, vg, x, tptr, yb);
  proj_gemm_kernel<<<dim3(2, 256), 256, 0, stream>>>(yb, wpb, b_proj, (float*)d_out);
}
